// DecomposedMaSA_80272938762742
// MI455X (gfx1250) — compile-verified, matrix operand order checked by reading, not run on hardware
//
#include <hip/hip_runtime.h>

// ---------------- problem constants ----------------
#define BATCH 8
#define CCH   128          // channels
#define HH    128
#define WW    128
#define HWP   (HH * WW)    // 16384
#define HEADS 4
#define DHEAD 32           // CCH / HEADS

typedef __bf16 bf16_t;
typedef __attribute__((ext_vector_type(16))) __bf16 bf16x16;
typedef __attribute__((ext_vector_type(8)))  float  v8f;
typedef __attribute__((ext_vector_type(4)))  unsigned int u32x4;
typedef __attribute__((ext_vector_type(8)))  int i32x8;
typedef __attribute__((ext_vector_type(4)))  int i32x4;

static __device__ __forceinline__ v8f vzero8() {
    v8f v;
#pragma unroll
    for (int i = 0; i < 8; ++i) v[i] = 0.0f;
    return v;
}

// Pack 4 floats -> 4 contiguous bf16 (one 8-byte LDS store).
static __device__ __forceinline__ void store4_bf16(bf16_t* dst, float a, float b,
                                                   float c, float d) {
    union { bf16_t h[4]; uint2 u; } pk;
    pk.h[0] = (bf16_t)a; pk.h[1] = (bf16_t)b; pk.h[2] = (bf16_t)c; pk.h[3] = (bf16_t)d;
    *reinterpret_cast<uint2*>(dst) = pk.u;
}

// ---------------------------------------------------------------------------
// Tensor Data Mover: async 2D tile load (f32 elements) global -> LDS.
// Builds the D# per CDNA5 ISA ch.8: rows of `width` contiguous elements,
// `rows` lines, line stride `stride` (elements). LDS receives rows packed
// consecutively: stage[row][x].
// Issue from ONE wave only; follow with s_wait_tensorcnt + workgroup barrier.
// ---------------------------------------------------------------------------
static __device__ __forceinline__ void tdm_load_2d_f32(unsigned lds_addr,
                                                       const float* gptr,
                                                       unsigned width, unsigned rows,
                                                       unsigned long long stride) {
    unsigned long long ga = (unsigned long long)(uintptr_t)gptr;
    u32x4 g0;
    g0[0] = 1u;                                          // count=1 (valid user D#)
    g0[1] = lds_addr;                                    // lds_addr     [63:32]
    g0[2] = (unsigned)(ga & 0xFFFFFFFFu);                // global_addr  [95:64]
    g0[3] = (unsigned)((ga >> 32) & 0x1FFFFFFu)          // global_addr  [120:96]
          | (2u << 30);                                  // type=2 ("image") [127:126]
    i32x8 g1;
    g1[0] = (int)(2u << 16);                             // data_size=2 (4B), mask=0
    g1[1] = (int)((width & 0xFFFFu) << 16);              // tensor_dim0 lo16 @ [63:48]
    g1[2] = (int)(((width >> 16) & 0xFFFFu)              // tensor_dim0 hi16
          | ((rows & 0xFFFFu) << 16));                   // tensor_dim1 lo16
    g1[3] = (int)(((rows >> 16) & 0xFFFFu)               // tensor_dim1 hi16
          | ((width & 0xFFFFu) << 16));                  // tile_dim0 @ [127:112]
    g1[4] = (int)(rows & 0xFFFFu);                       // tile_dim1; tile_dim2=0
    g1[5] = (int)(unsigned)(stride & 0xFFFFFFFFull);     // tensor_dim0_stride[31:0]
    g1[6] = (int)(unsigned)((stride >> 32) & 0xFFFFull); // stride hi16; dim1_stride lo=0
    g1[7] = 0;
    i32x4 z4 = {0, 0, 0, 0};
#if defined(__clang_major__) && (__clang_major__ >= 23)
    i32x8 z8 = {0, 0, 0, 0, 0, 0, 0, 0};
    __builtin_amdgcn_tensor_load_to_lds(g0, g1, z4, z4, z8, 0);
#else
    __builtin_amdgcn_tensor_load_to_lds(g0, g1, z4, z4, 0);
#endif
}

// Low 32 bits of a flat LDS pointer are the LDS byte offset (ISA 10.2 aperture map).
static __device__ __forceinline__ unsigned lds_byte_addr(const void* p) {
    return (unsigned)(uintptr_t)p;
}

// Load a 16-bit WMMA fragment (A-style; B uses the same per-lane K mapping with
// lane = column) from LDS stored row-major with contiguous K.
// Per ISA 7.12.2 (16-bit A 16x32): lanes 0-15 hold K 0-7 & 16-23,
// lanes 16-31 hold K 8-15 & 24-31, each as two contiguous 16B chunks.
static __device__ __forceinline__ bf16x16 load_frag_k32(const bf16_t* p, int half) {
    union { uint4 u[2]; bf16x16 v; } t;
    const uint4* q = reinterpret_cast<const uint4*>(p);
    t.u[0] = q[half];       // elements 0..7  -> K = 8*half .. 8*half+7
    t.u[1] = q[2 + half];   // elements 8..15 -> K = 16+8*half .. 16+8*half+7
    return t.v;
}

// ---------------------------------------------------------------------------
// 1x1 conv as GEMM:  Y[b, m, pix] = sum_k Wt[m, k] * X[b, k, pix]
// Workgroup: 8 waves -> 64 (out-ch) x 128 (pixels) tile.
// A/B tiles fetched by the Tensor Data Mover into f32 LDS stages, then
// converted to bf16 fragment layout with packed B64 stores.
// grid = (HWP/128, BATCH, Cout/64)
// ---------------------------------------------------------------------------
__global__ __launch_bounds__(256)
void conv1x1_wmma_kernel(const float* __restrict__ X, const float* __restrict__ Wt,
                         float* __restrict__ Y, int Cin) {
    __shared__ __attribute__((aligned(16))) float  sA[64 * 32];    // TDM stage [m][k]
    __shared__ __attribute__((aligned(16))) float  sB[32 * 128];   // TDM stage [k][n]
    __shared__ __attribute__((aligned(16))) bf16_t lA[64 * 32];    // [m][k]
    __shared__ __attribute__((aligned(16))) bf16_t lB[128 * 32];   // [n][k] (transposed)

    const int b    = blockIdx.y;
    const int m0   = blockIdx.z * 64;
    const int n0   = blockIdx.x * 128;
    const int Cout = gridDim.z * 64;

    const float* xb = X + (size_t)b * Cin * HWP;
    float*       yb = Y + (size_t)b * Cout * HWP;

    const int tid  = threadIdx.x;
    const int lane = tid & 31;
    const int wave = tid >> 5;       // 0..7
    const int wm   = wave >> 1;      // 0..3 : 16-row sub-tile
    const int wn   = wave & 1;       // 0..1 : 64-col half
    const int nn   = lane & 15;
    const int half = lane >> 4;

    v8f acc[4];
#pragma unroll
    for (int t = 0; t < 4; ++t) acc[t] = vzero8();

    for (int k0 = 0; k0 < Cin; k0 += 32) {
        // --- async TDM fetch of both tiles (wave 0 issues) ---
        if (wave == 0) {
            tdm_load_2d_f32(lds_byte_addr(sA), Wt + (size_t)m0 * Cin + k0,
                            32, 64, (unsigned long long)Cin);
            tdm_load_2d_f32(lds_byte_addr(sB), xb + (size_t)k0 * HWP + n0,
                            128, 32, (unsigned long long)HWP);
            __builtin_amdgcn_s_wait_tensorcnt(0);
        }
        __syncthreads();

        // --- convert stages to bf16 fragment layout ---
        // A: same layout, packed 4-wide stores. 2048 elems = 512 quads.
        for (int i = tid; i < 512; i += 256) {
            const float* s = &sA[i * 4];
            store4_bf16(&lA[i * 4], s[0], s[1], s[2], s[3]);
        }
        // B: transpose [k][n] -> [n][k], 4 consecutive k packed per store.
        for (int i = tid; i < 1024; i += 256) {
            int n = i >> 3, kb = (i & 7) * 4;
            store4_bf16(&lB[n * 32 + kb],
                        sB[(kb + 0) * 128 + n], sB[(kb + 1) * 128 + n],
                        sB[(kb + 2) * 128 + n], sB[(kb + 3) * 128 + n]);
        }
        __syncthreads();

        bf16x16 afrag = load_frag_k32(&lA[(wm * 16 + nn) * 32], half);
#pragma unroll
        for (int t = 0; t < 4; ++t) {
            bf16x16 bfrag = load_frag_k32(&lB[(wn * 64 + t * 16 + nn) * 32], half);
            acc[t] = __builtin_amdgcn_wmma_f32_16x16x32_bf16(
                false, afrag, false, bfrag, (short)0, acc[t], false, false);
        }
        __syncthreads();
    }

    // C/D layout: lane -> (m = r + 8*half, n = lane&15); stores coalesced over lanes.
    const int mh = half * 8;
#pragma unroll
    for (int t = 0; t < 4; ++t) {
#pragma unroll
        for (int r = 0; r < 8; ++r) {
            int m = m0 + wm * 16 + mh + r;
            int n = n0 + wn * 64 + t * 16 + nn;
            yb[(size_t)m * HWP + n] = acc[t][r];
        }
    }
}

// ---------------------------------------------------------------------------
// Axis-decomposed attention over a 128-long sequence, d = 32.
// One workgroup per (b, head, fixed coordinate f).
//   axis 0: attend along W within row i=f   (pos stride 1,  base f*W) -> TDM staging
//   axis 1: attend along H within col w=f   (pos stride W,  base f)   -> manual staging
// S = softmax(scale*Q K^T) * decay ; O = S V ; out written (=) or (+=).
// grid = (128, HEADS, BATCH)
// ---------------------------------------------------------------------------
__global__ __launch_bounds__(256)
void axis_attn_wmma_kernel(const float* __restrict__ qkv, float* __restrict__ out,
                           const float* __restrict__ gamma_p, int axis, int accum) {
    __shared__ __attribute__((aligned(16))) bf16_t Qlds[128 * 32];   // [pos][d], pre-scaled
    __shared__ __attribute__((aligned(16))) bf16_t Klds[128 * 32];   // [pos][d]
    __shared__ __attribute__((aligned(16))) bf16_t Vlds[32 * 128];   // [d][pos] (transposed)
    __shared__ __attribute__((aligned(16))) bf16_t Plds[128 * 128];  // softmax*decay, bf16

    const int f    = blockIdx.x;
    const int head = blockIdx.y;
    const int b    = blockIdx.z;

    const int    stride_pos = (axis == 0) ? 1 : WW;
    const size_t base       = (axis == 0) ? (size_t)f * WW : (size_t)f;

    const float* qb = qkv + ((size_t)b * 3 * CCH + head * DHEAD) * HWP + base;
    float*       yb = out + ((size_t)b * CCH + head * DHEAD) * HWP + base;

    const int tid  = threadIdx.x;
    const int lane = tid & 31;
    const int wave = tid >> 5;     // 0..7 : 16-query-row stripe
    const int nn   = lane & 15;
    const int half = lane >> 4;

    const float gam   = *gamma_p;
    const float scale = 0.17677669529663687f;   // 32^-0.5

    if (axis == 0) {
        // ---- TDM staging: Q,K then V through f32 stages overlaid on Plds ----
        float* st0 = reinterpret_cast<float*>(Plds);          // 16 KB: [d][pos]
        float* st1 = st0 + 4096;                              // 16 KB

        if (wave == 0) {
            tdm_load_2d_f32(lds_byte_addr(st0), qb, 128, 32, HWP);
            tdm_load_2d_f32(lds_byte_addr(st1), qb + (size_t)CCH * HWP, 128, 32, HWP);
            __builtin_amdgcn_s_wait_tensorcnt(0);
        }
        __syncthreads();
        // transpose-convert [d][pos] -> [pos][d], 4 consecutive d packed per store
        for (int i = tid; i < 1024; i += 256) {
            int p = i >> 3, db = (i & 7) * 4;
            store4_bf16(&Qlds[p * 32 + db],
                        st0[(db + 0) * 128 + p] * scale, st0[(db + 1) * 128 + p] * scale,
                        st0[(db + 2) * 128 + p] * scale, st0[(db + 3) * 128 + p] * scale);
            store4_bf16(&Klds[p * 32 + db],
                        st1[(db + 0) * 128 + p], st1[(db + 1) * 128 + p],
                        st1[(db + 2) * 128 + p], st1[(db + 3) * 128 + p]);
        }
        __syncthreads();
        if (wave == 0) {
            tdm_load_2d_f32(lds_byte_addr(st0), qb + (size_t)(2 * CCH) * HWP, 128, 32, HWP);
            __builtin_amdgcn_s_wait_tensorcnt(0);
        }
        __syncthreads();
        // V stage layout [d][pos] == Vlds layout: straight packed convert
        for (int i = tid; i < 1024; i += 256) {
            const float* s = &st0[i * 4];
            store4_bf16(&Vlds[i * 4], s[0], s[1], s[2], s[3]);
        }
        __syncthreads();
    } else {
        // ---- manual staging (512B-strided gather; no dense-tile shape) ----
        for (int i = tid; i < 128 * 32; i += 256) {
            int dd = i >> 7;     // 0..31
            int p  = i & 127;    // 0..127
            size_t g = (size_t)dd * HWP + (size_t)p * stride_pos;
            float qv = qb[g];
            float kv = qb[g + (size_t)CCH * HWP];
            float vv = qb[g + (size_t)(2 * CCH) * HWP];
            Qlds[p * 32 + dd]  = (bf16_t)(qv * scale);
            Klds[p * 32 + dd]  = (bf16_t)kv;
            Vlds[dd * 128 + p] = (bf16_t)vv;
        }
        __syncthreads();
    }

    // ---- S = (scale*Q) K^T : K-dim = 32 -> ONE wmma per 16x16 tile ----
    v8f accS[8];
#pragma unroll
    for (int t = 0; t < 8; ++t) accS[t] = vzero8();

    bf16x16 qf = load_frag_k32(&Qlds[(wave * 16 + nn) * 32], half);
#pragma unroll
    for (int t = 0; t < 8; ++t) {
        bf16x16 kf = load_frag_k32(&Klds[(t * 16 + nn) * 32], half);
        accS[t] = __builtin_amdgcn_wmma_f32_16x16x32_bf16(
            false, qf, false, kf, (short)0, accS[t], false, false);
    }

    // ---- register softmax over each row of the wave's 16-row stripe ----
    // Row m = r + 8*half lives in lanes {16*half .. 16*half+15}, col = 16t + nn.
#pragma unroll
    for (int r = 0; r < 8; ++r) {
        float mx = -3.0e38f;
#pragma unroll
        for (int t = 0; t < 8; ++t) mx = fmaxf(mx, accS[t][r]);
#pragma unroll
        for (int m = 1; m < 16; m <<= 1) mx = fmaxf(mx, __shfl_xor(mx, m, 32));

        float pv[8];
        float sum = 0.0f;
#pragma unroll
        for (int t = 0; t < 8; ++t) { pv[t] = __expf(accS[t][r] - mx); sum += pv[t]; }
#pragma unroll
        for (int m = 1; m < 16; m <<= 1) sum += __shfl_xor(sum, m, 32);
        float inv = 1.0f / sum;

        int j = wave * 16 + r + 8 * half;      // query index
#pragma unroll
        for (int t = 0; t < 8; ++t) {
            int z = t * 16 + nn;               // key index
            float dec = __expf(-gam * fabsf((float)(j - z)));
            Plds[j * 128 + z] = (bf16_t)(pv[t] * inv * dec);
        }
    }
    __syncthreads();

    // ---- O = P V : M=16 rows/wave, N=32 (2 tiles), K=128 (4 steps) ----
    v8f accO[2];
    accO[0] = vzero8();
    accO[1] = vzero8();
#pragma unroll
    for (int k0 = 0; k0 < 128; k0 += 32) {
        bf16x16 af = load_frag_k32(&Plds[(wave * 16 + nn) * 128 + k0], half);
#pragma unroll
        for (int nt = 0; nt < 2; ++nt) {
            bf16x16 bf = load_frag_k32(&Vlds[(nt * 16 + nn) * 128 + k0], half);
            accO[nt] = __builtin_amdgcn_wmma_f32_16x16x32_bf16(
                false, af, false, bf, (short)0, accO[nt], false, false);
        }
    }

    // ---- write out[b, head*32+dim, pos j] ----
#pragma unroll
    for (int nt = 0; nt < 2; ++nt) {
        int dim = nt * 16 + nn;
#pragma unroll
        for (int r = 0; r < 8; ++r) {
            int j = wave * 16 + r + 8 * half;
            size_t idx = (size_t)dim * HWP + (size_t)j * stride_pos;
            if (accum) yb[idx] += accO[nt][r];
            else       yb[idx]  = accO[nt][r];
        }
    }
}

// ---------------------------------------------------------------------------
// Depthwise 3x3, SAME padding. One thread per output element.
// ---------------------------------------------------------------------------
__global__ __launch_bounds__(256)
void dwconv3x3_kernel(const float* __restrict__ in, const float* __restrict__ wdw,
                      float* __restrict__ outp) {
    int gid = blockIdx.x * 256 + threadIdx.x;
    int w  = gid & (WW - 1);
    int h  = (gid >> 7) & (HH - 1);
    int bc = gid >> 14;              // b*C + c
    int c  = bc & (CCH - 1);

    const float* p  = in + (size_t)bc * HWP;
    const float* wk = wdw + c * 9;

    float s = 0.0f;
#pragma unroll
    for (int ky = 0; ky < 3; ++ky) {
        int hh = h + ky - 1;
        if (hh < 0 || hh >= HH) continue;
#pragma unroll
        for (int kx = 0; kx < 3; ++kx) {
            int ww2 = w + kx - 1;
            if (ww2 < 0 || ww2 >= WW) continue;
            s += p[hh * WW + ww2] * wk[ky * 3 + kx];
        }
    }
    outp[gid] = s;
}

// ---------------------------------------------------------------------------
// Pipeline: qkv GEMM -> W-axis attn (=) -> H-axis attn (+=) -> dw3x3 -> proj GEMM
// Workspace: qkv 192 MB + dw temp 64 MB = 256 MB from d_ws.
// ---------------------------------------------------------------------------
extern "C" void kernel_launch(void* const* d_in, const int* in_sizes, int n_in,
                              void* d_out, int out_size, void* d_ws, size_t ws_size,
                              hipStream_t stream) {
    (void)in_sizes; (void)n_in; (void)out_size; (void)ws_size;
    const float* x      = (const float*)d_in[0];
    const float* w_qkv  = (const float*)d_in[1];
    const float* w_proj = (const float*)d_in[2];
    const float* w_dw   = (const float*)d_in[3];
    const float* gamma  = (const float*)d_in[4];
    float* out = (float*)d_out;

    float* qkv = (float*)d_ws;                               // [B, 3C, H, W]
    float* tmp = qkv + (size_t)BATCH * 3 * CCH * HWP;        // [B, C, H, W]

    {   // qkv = w_qkv @ x     (Cout = 384)
        dim3 g(HWP / 128, BATCH, (3 * CCH) / 64);
        conv1x1_wmma_kernel<<<g, 256, 0, stream>>>(x, w_qkv, qkv, CCH);
    }
    {   // decomposed attention: W-axis writes, H-axis accumulates
        dim3 g(WW, HEADS, BATCH);
        axis_attn_wmma_kernel<<<g, 256, 0, stream>>>(qkv, out, gamma, 0, 0);
        axis_attn_wmma_kernel<<<g, 256, 0, stream>>>(qkv, out, gamma, 1, 1);
    }
    {   // LePE depthwise 3x3 into tmp
        int total = BATCH * CCH * HWP;
        dwconv3x3_kernel<<<total / 256, 256, 0, stream>>>(out, w_dw, tmp);
    }
    {   // out = w_proj @ tmp  (Cout = 128)
        dim3 g(HWP / 128, BATCH, CCH / 64);
        conv1x1_wmma_kernel<<<g, 256, 0, stream>>>(tmp, w_proj, out, CCH);
    }
}
